// DCNv2_31705448579230
// MI455X (gfx1250) — matrix operand order repaired and checked by reading, not compiled, not run on hardware
//
#include <hip/hip_runtime.h>

typedef __attribute__((ext_vector_type(16))) _Float16     v16h;
typedef __attribute__((ext_vector_type(8)))  float        v8f;
typedef __attribute__((ext_vector_type(4)))  unsigned int v4u;

#define DIM_B 8
#define DIM_H 64
#define DIM_W 64
#define DIM_C 128
#define DIM_F 256
#define NK    9
#define MK    1152              // 9 * 128  (GEMM K)
#define MM    32768             // 8*64*64  (GEMM M)
#define KSTEPS 36               // 1152 / 32

// ---------------------------------------------------------------------------
// Kernel 1: deformable bilinear sampling -> im2col matrix A [M=32768, K=1152] f16
// One block per output pixel m=(b,oh,ow); 128 threads = channels.
// Replicates reference exactly: padded coords, clamp-to-[0,65], floor,
// weight/sample pairing (f01 = (y1,x0) with weight (1-ly)*lx, etc.).
// ---------------------------------------------------------------------------
__global__ __launch_bounds__(128) void dcn_sample(
    const float* __restrict__ x, const float* __restrict__ off,
    const float* __restrict__ mod, _Float16* __restrict__ A) {
  const int m = blockIdx.x;
  const int c = threadIdx.x;
  const int b  = m >> 12;
  const int oh = (m >> 6) & 63;
  const int ow = m & 63;
  _Float16* arow = A + (size_t)m * MK + c;

#pragma unroll
  for (int t = 0; t < NK; ++t) {
    const int ky = t / 3, kx = t % 3;
    const int r = 3 * oh + ky;          // expanded row
    const int h = r & 63;               // source pixel row (reference's reshape)
    const int g = r >> 6;               // tap-row group
    const int k = 3 * g + kx;           // tap index into NK axis

    const int base = (b * DIM_H + h) * DIM_W + ow;
    const float oy = off[base * (2 * NK) + k];
    const float ox = off[base * (2 * NK) + NK + k];
    const float mv = mod[base * NK + k];

    // padded-frame sampling point: pixel center (h+1, ow+1) + tap offset
    const float py = oy + (float)(h + 1) + (float)(k / 3 - 1);
    const float px = ox + (float)(ow + 1) + (float)(k % 3 - 1);

    const float y0f = floorf(py), x0f = floorf(px);
    const float y0 = fminf(fmaxf(y0f,        0.f), 65.f);
    const float y1 = fminf(fmaxf(y0f + 1.f,  0.f), 65.f);
    const float x0 = fminf(fmaxf(x0f,        0.f), 65.f);
    const float x1 = fminf(fmaxf(x0f + 1.f,  0.f), 65.f);
    const float ly = fminf(fmaxf(py, 0.f), 65.f) - y0;
    const float lx = fminf(fmaxf(px, 0.f), 65.f) - x0;

    const int iy0 = (int)y0, iy1 = (int)y1, ix0 = (int)x0, ix1 = (int)x1;

    auto fetch = [&](int yy, int xx) -> float {
      if (yy < 1 || yy > DIM_H || xx < 1 || xx > DIM_W) return 0.f;   // zero pad
      return x[((size_t)(b * DIM_H + (yy - 1)) * DIM_W + (xx - 1)) * DIM_C + c];
    };
    const float f00 = fetch(iy0, ix0);
    const float f01 = fetch(iy1, ix0);
    const float f10 = fetch(iy0, ix1);
    const float f11 = fetch(iy1, ix1);

    const float val = ((1.f - ly) * (1.f - lx) * f00 + (1.f - ly) * lx * f01 +
                       ly * (1.f - lx) * f10 + ly * lx * f11) * mv;
    arow[t * DIM_C] = (_Float16)val;
  }
}

// ---------------------------------------------------------------------------
// Kernel 2: pack weights [1152, 256] f32 into WMMA B-fragment order (f16).
// Fragment (s, nt): 32x16 tile, per-lane 8 dwords:
//   lane<16 : n = nt*16+lane,    vgpr v -> K = s*32 + 2v, 2v+1
//   lane>=16: n = nt*16+lane-16, vgpr v -> K = s*32 + 16 + 2v, +2v+1
// Bp flat layout: [s][nt][lane][v] dwords  (2 halves per dword, low = even K)
// ---------------------------------------------------------------------------
__global__ void dcn_pack_w(const float* __restrict__ Kw, _Float16* __restrict__ Bp) {
  const int idx = blockIdx.x * blockDim.x + threadIdx.x;   // dword index
  const int total = KSTEPS * 16 * 32 * 8;
  if (idx >= total) return;
  const int v    = idx & 7;
  const int lane = (idx >> 3) & 31;
  const int nt   = (idx >> 8) & 15;
  const int s    = idx >> 12;
  const int n = nt * 16 + (lane & 15);
  const int k = s * 32 + ((lane >> 4) << 4) + 2 * v;
  Bp[(size_t)idx * 2]     = (_Float16)Kw[(size_t)k * DIM_F + n];
  Bp[(size_t)idx * 2 + 1] = (_Float16)Kw[(size_t)(k + 1) * DIM_F + n];
}

// ---------------------------------------------------------------------------
// Kernel 3: WMMA GEMM  out[M,256] = A[M,1152] x W'[1152,256] + bias
// 4 waves/block; wave w computes 32(M) x 64(N) at (m0 = 32*blockIdx, n0 = 64*w).
// A fragment (matches packed B): lane<16 -> K {0..15},
// lane>=16 -> K {16..31} of the 32-wide K-step.
// ---------------------------------------------------------------------------
__device__ __forceinline__ v16h load_frag(const _Float16* p0, const _Float16* p1) {
  union { v4u u[2]; v16h v; } r;
  r.u[0] = *(const v4u*)p0;
  r.u[1] = *(const v4u*)p1;
  return r.v;
}

__global__ __launch_bounds__(128) void dcn_gemm(
    const _Float16* __restrict__ A, const _Float16* __restrict__ Bp,
    const float* __restrict__ bias, float* __restrict__ out) {
  const int lane = threadIdx.x & 31;
  const int wave = threadIdx.x >> 5;
  const int m0 = blockIdx.x * 32;
  const int n0 = wave * 64;

  const int mrow  = lane & 15;
  const int khalf = lane >> 4;                       // 0 or 1
  const _Float16* a0 = A + (size_t)(m0 + mrow) * MK + khalf * 16;
  const _Float16* a1 = a0 + (size_t)16 * MK;
  const _Float16* bbase = Bp + ((size_t)(wave * 4) * 32 + lane) * 16;

  v8f acc[2][4] = {};

  for (int s = 0; s < KSTEPS; ++s) {
    const v16h af0 = load_frag(a0, a0 + 8);
    const v16h af1 = load_frag(a1, a1 + 8);
    const _Float16* bp = bbase + (size_t)s * (16 * 32 * 16);
#pragma unroll
    for (int j = 0; j < 4; ++j) {
      const v16h bf = load_frag(bp + j * 512, bp + j * 512 + 8);
      acc[0][j] = __builtin_amdgcn_wmma_f32_16x16x32_f16(
          false, af0, false, bf, (short)0, acc[0][j], false, false);
      acc[1][j] = __builtin_amdgcn_wmma_f32_16x16x32_f16(
          false, af1, false, bf, (short)0, acc[1][j], false, false);
    }
    a0 += 32;
    a1 += 32;
  }

  // Epilogue: f32 C/D layout -> lanes 0-15: M=g, N=lane; lanes 16-31: M=8+g.
  const int nlane = lane & 15;
  const int mhi   = (lane >> 4) * 8;
#pragma unroll
  for (int mt = 0; mt < 2; ++mt) {
#pragma unroll
    for (int j = 0; j < 4; ++j) {
      const int col = n0 + j * 16 + nlane;
      const float bv = bias[col];
#pragma unroll
      for (int g = 0; g < 8; ++g) {
        const int row = m0 + mt * 16 + mhi + g;
        out[(size_t)row * DIM_F + col] = acc[mt][j][g] + bv;
      }
    }
  }
}

// ---------------------------------------------------------------------------
extern "C" void kernel_launch(void* const* d_in, const int* in_sizes, int n_in,
                              void* d_out, int out_size, void* d_ws, size_t ws_size,
                              hipStream_t stream) {
  (void)in_sizes; (void)n_in; (void)out_size; (void)ws_size;
  const float* x    = (const float*)d_in[0];
  const float* off  = (const float*)d_in[1];
  const float* mod  = (const float*)d_in[2];
  const float* Kw   = (const float*)d_in[3];
  const float* bias = (const float*)d_in[4];
  float* out = (float*)d_out;

  _Float16* A  = (_Float16*)d_ws;                                   // 75,497,472 B
  _Float16* Bp = (_Float16*)((char*)d_ws + (size_t)MM * MK * 2);    // +589,824 B

  dcn_sample<<<MM, 128, 0, stream>>>(x, off, mod, A);
  const int packTotal = KSTEPS * 16 * 32 * 8;
  dcn_pack_w<<<(packTotal + 255) / 256, 256, 0, stream>>>(Kw, Bp);
  dcn_gemm<<<MM / 32, 128, 0, stream>>>(A, Bp, bias, out);
}
